// MultiHeadAttention_39487929319891
// MI455X (gfx1250) — matrix operand order repaired and checked by reading, not compiled, not run on hardware
//
#include <hip/hip_runtime.h>
#include <hip/hip_bf16.h>
#include <stdint.h>

// ---------------- problem constants ----------------
#define B_   2
#define L_   2048
#define D_   2048
#define NH_  16
#define HD_  128
#define M_   (B_ * L_)               // 4096 tokens
#define SCALE_ 0.08838834764831845f  // HD^-0.5

typedef __bf16 bf16;
typedef __attribute__((ext_vector_type(16))) __bf16 v16bf;
typedef __attribute__((ext_vector_type(8)))  __bf16 v8bf;
typedef __attribute__((ext_vector_type(8)))  float  v8f;
typedef __attribute__((address_space(3))) uint16_t lds_u16;

static __device__ __forceinline__ v8f wmma_bf16(v16bf a, v16bf b, v8f c) {
  // D = A(16x32 bf16) * B(32x16 bf16) + C(16x16 f32)
  return __builtin_amdgcn_wmma_f32_16x16x32_bf16(false, a, false, b, (short)0, c, false, false);
}

static __device__ __forceinline__ uint16_t f2bf(float f) {
  uint32_t u = __float_as_uint(f);
  u += 0x7fffu + ((u >> 16) & 1u);   // round-to-nearest-even
  return (uint16_t)(u >> 16);
}

static __device__ __forceinline__ v16bf combine8(v8bf lo, v8bf hi) {
  v16bf a;
#pragma unroll
  for (int i = 0; i < 8; ++i) { a[i] = lo[i]; a[i + 8] = hi[i]; }
  return a;
}

// Async DMA: 16 bytes/lane global -> LDS, tracked by ASYNCcnt (CDNA5 path).
static __device__ __forceinline__ void async_g2l_b128(uint32_t lds_off, const void* gaddr) {
  asm volatile("global_load_async_to_lds_b128 %0, %1, off"
               :: "v"(lds_off), "v"(gaddr) : "memory");
}
static __device__ __forceinline__ void wait_async_le1() {
  asm volatile("s_wait_asynccnt 1" ::: "memory");
}
static __device__ __forceinline__ void wait_async_le0() {
  asm volatile("s_wait_asynccnt 0" ::: "memory");
}

// ---------------- fp32 -> bf16 convert ----------------
__global__ void cvt_f32_bf16(const float* __restrict__ in, uint16_t* __restrict__ out, int n) {
  int i = blockIdx.x * blockDim.x + threadIdx.x;
  int stride = gridDim.x * blockDim.x;
  for (; i < n; i += stride) out[i] = f2bf(in[i]);
}

// ---------------- WMMA GEMM: C[m,n] = sum_k A[m,k] * W[n,k] + bias[n] ----------------
// Block = 256 threads (8 waves) -> 32x512 tile. A panel (32x32/K-step) staged in LDS
// via async DMA (double-buffered), shared by all 8 waves. Wave tile = 32x64:
// 8 WMMAs per K-step with A-fragment reuse x4 and B-fragment reuse x2.
// MODE 0: bf16 out row-major [m, n]
// MODE 1: f32  out row-major [m, n]            (final projection -> d_out)
// MODE 2: bf16 out scattered to Vt[b][h][d][l] (per-head transposed V)
template <int MODE>
__global__ __launch_bounds__(256) void gemm32(const uint16_t* __restrict__ Au,
                                              const uint16_t* __restrict__ Wu,
                                              const float* __restrict__ bias,
                                              void* __restrict__ outp) {
  __shared__ uint16_t sA[2][32][32];            // 2 x 2KB double buffer
  const int lane = threadIdx.x & 31;
  const int w    = threadIdx.x >> 5;
  const int tn8  = blockIdx.x & 3;              // N/512 = 4 column groups
  const int tm32 = blockIdx.x >> 2;             // M/32  = 128 row tiles
  const int nlo  = lane & 15;
  const int lg   = lane >> 4;                   // wave32 half
  const int klo  = lg ? 16 : 0;                 // A-fragment K sub-base
  const int kh   = lg ? 16 : 0;                 // B-fragment K half
  const int mb   = lg ? 8 : 0;                  // C-fragment row base

  const bf16* A = (const bf16*)Au;
  const bf16* W = (const bf16*)Wu;

  // A-panel DMA addressing: 128 lanes (waves 0..3) x 16B cover 32x32 bf16
  const int cid = threadIdx.x & 127;
  const int am  = cid >> 2;                     // 0..31 panel row
  const int ak  = (cid & 3) * 8;                // 0,8,16,24 (bf16 elems)
  const bf16* abase = A + (size_t)(tm32 * 32 + am) * D_ + ak;
  uint32_t ldsoff[2];
#pragma unroll
  for (int buf = 0; buf < 2; ++buf)
    ldsoff[buf] = (uint32_t)(size_t)(lds_u16*)&sA[buf][am][ak];

  const bf16* wcol[4];
#pragma unroll
  for (int j = 0; j < 4; ++j)
    wcol[j] = W + (size_t)(tn8 * 512 + w * 64 + j * 16 + nlo) * D_;

  if (w < 4) async_g2l_b128(ldsoff[0], abase);  // prologue: panel kk=0

  v8f c[2][4] = {};
  int cur = 0;
  for (int kk = 0; kk < D_; kk += 32) {
    if (w < 4) {
      if (kk + 32 < D_) {                       // prefetch next panel, wait for current
        async_g2l_b128(ldsoff[cur ^ 1], abase + kk + 32);
        wait_async_le1();
      } else {
        wait_async_le0();
      }
    }
    __syncthreads();                            // panel visible to all 8 waves

    v16bf a[2];
#pragma unroll
    for (int mt = 0; mt < 2; ++mt) {
      const int m = mt * 16 + nlo;
      v8bf lo = *(const v8bf*)&sA[cur][m][klo];
      v8bf hi = *(const v8bf*)&sA[cur][m][klo + 8];
      a[mt] = combine8(lo, hi);
    }
#pragma unroll
    for (int j = 0; j < 4; ++j) {
      v16bf bfrag = *(const v16bf*)(wcol[j] + kk + kh);
      c[0][j] = wmma_bf16(a[0], bfrag, c[0][j]);
      c[1][j] = wmma_bf16(a[1], bfrag, c[1][j]);
    }
    __syncthreads();                            // all reads done before buffer reuse
    cur ^= 1;
  }

#pragma unroll
  for (int mt = 0; mt < 2; ++mt) {
#pragma unroll
    for (int j = 0; j < 4; ++j) {
      const int ncol = tn8 * 512 + w * 64 + j * 16 + nlo;
      const float bv = bias[ncol];
#pragma unroll
      for (int r = 0; r < 8; ++r) {
        const int mg = tm32 * 32 + mt * 16 + mb + r;
        const float val = c[mt][j][r] + bv;
        if (MODE == 0) {
          ((uint16_t*)outp)[(size_t)mg * D_ + ncol] = f2bf(val);
        } else if (MODE == 1) {
          ((float*)outp)[(size_t)mg * D_ + ncol] = val;
        } else {
          const int bb = mg >> 11, ll = mg & (L_ - 1);
          const int hh = ncol >> 7, dd = ncol & (HD_ - 1);
          ((uint16_t*)outp)[((size_t)(bb * NH_ + hh) * HD_ + dd) * L_ + ll] = f2bf(val);
        }
      }
    }
  }
}

// ---------------- Flash attention: one wave per 16-query tile per (b,h) ----------------
// Q,K layout: [b, l, h, d] bf16 ; Vt layout: [b, h, d, l] bf16 ; O out: [b, l, h*HD+d] bf16
__global__ __launch_bounds__(128) void attn_fa(const uint16_t* __restrict__ Qu,
                                               const uint16_t* __restrict__ Ku,
                                               const uint16_t* __restrict__ Vtu,
                                               uint16_t* __restrict__ Ou) {
  __shared__ uint16_t sP[4][16][32];            // per-wave P transpose staging
  const int lane = threadIdx.x & 31;
  const int w    = threadIdx.x >> 5;
  const int wid  = blockIdx.x * 4 + w;
  const int qt   = wid & (L_ / 16 - 1);
  const int bh   = wid >> 7;
  const int b    = bh >> 4;
  const int h    = bh & 15;
  const int q0   = qt * 16;
  const int nlo  = lane & 15;
  const int lg   = lane >> 4;
  const int klo  = lg ? 16 : 0;
  const int kh   = lg ? 16 : 0;
  const int mb   = lg ? 8 : 0;

  const bf16* Q  = (const bf16*)Qu;
  const bf16* K  = (const bf16*)Ku;
  const bf16* Vt = (const bf16*)Vtu;

  // resident Q fragments: 4 chunks of K-dim (HD = 4*32)
  const bf16* qrow = Q + ((size_t)(b * L_ + q0 + nlo) * NH_ + h) * HD_;
  v16bf aq[4];
#pragma unroll
  for (int ck = 0; ck < 4; ++ck) {
    v8bf lo = *(const v8bf*)(qrow + ck * 32 + klo);
    v8bf hi = *(const v8bf*)(qrow + ck * 32 + klo + 8);
    aq[ck] = combine8(lo, hi);
  }

  float mrow[8], lrow[8];
  v8f o[8] = {};
#pragma unroll
  for (int r = 0; r < 8; ++r) { mrow[r] = -1e30f; lrow[r] = 0.f; }

  const int nblk = (q0 + 16 + 31) >> 5;         // 32-key blocks
  for (int blk = 0; blk < nblk; ++blk) {
    const int k0 = blk * 32;
    v8f s[2];
#pragma unroll
    for (int t = 0; t < 2; ++t) {
      const bf16* krow = K + ((size_t)(b * L_ + k0 + t * 16 + nlo) * NH_ + h) * HD_;
      v8f acc = {};
#pragma unroll
      for (int ck = 0; ck < 4; ++ck) {
        v16bf bk = *(const v16bf*)(krow + ck * 32 + kh);
        acc = wmma_bf16(aq[ck], bk, acc);
      }
      s[t] = acc;
    }
    // scale + causal mask
#pragma unroll
    for (int t = 0; t < 2; ++t) {
      const int kcol = k0 + t * 16 + nlo;
#pragma unroll
      for (int r = 0; r < 8; ++r) {
        const float sv = s[t][r] * SCALE_;
        s[t][r] = (kcol <= q0 + mb + r) ? sv : -1e30f;
      }
    }
    // online softmax (row stats replicated across the 16 lanes of each half)
#pragma unroll
    for (int r = 0; r < 8; ++r) {
      float v = fmaxf(s[0][r], s[1][r]);
#pragma unroll
      for (int off = 1; off < 16; off <<= 1) v = fmaxf(v, __shfl_xor(v, off, 16));
      const float mnew = fmaxf(mrow[r], v);
      const float corr = __expf(mrow[r] - mnew);
      mrow[r] = mnew;
      const float p0 = __expf(s[0][r] - mnew);
      const float p1 = __expf(s[1][r] - mnew);
      float ps = p0 + p1;
#pragma unroll
      for (int off = 1; off < 16; off <<= 1) ps += __shfl_xor(ps, off, 16);
      lrow[r] = lrow[r] * corr + ps;
#pragma unroll
      for (int dt = 0; dt < 8; ++dt) o[dt][r] *= corr;
      const int m = mb + r;
      sP[w][m][nlo]      = f2bf(p0);            // C-layout -> LDS (row-major 16x32)
      sP[w][m][16 + nlo] = f2bf(p1);
    }
    asm volatile("s_wait_dscnt 0" ::: "memory");  // same-wave LDS stores -> loads
    v8bf plo = *(const v8bf*)&sP[w][nlo][klo];
    v8bf phi = *(const v8bf*)&sP[w][nlo][klo + 8];
    v16bf ap = combine8(plo, phi);
    // PV: O(16x128) += P(16x32) x V(32x128), B-fragments contiguous thanks to Vt
#pragma unroll
    for (int dt = 0; dt < 8; ++dt) {
      const bf16* vrow = Vt + ((size_t)(b * NH_ + h) * HD_ + dt * 16 + nlo) * L_ + k0 + kh;
      v16bf bv = *(const v16bf*)vrow;
      o[dt] = wmma_bf16(ap, bv, o[dt]);
    }
  }

  // epilogue: O /= l, store bf16 [b, q, h*HD + d]
#pragma unroll
  for (int r = 0; r < 8; ++r) {
    const float inv = 1.f / lrow[r];
    const int q = q0 + mb + r;
#pragma unroll
    for (int dt = 0; dt < 8; ++dt) {
      const int col = h * HD_ + dt * 16 + nlo;
      Ou[(size_t)(b * L_ + q) * D_ + col] = f2bf(o[dt][r] * inv);
    }
  }
}

// ---------------- host launcher ----------------
extern "C" void kernel_launch(void* const* d_in, const int* in_sizes, int n_in,
                              void* d_out, int out_size, void* d_ws, size_t ws_size,
                              hipStream_t stream) {
  const float* hs = (const float*)d_in[0];
  const float* Wq = (const float*)d_in[1];
  const float* bq = (const float*)d_in[2];
  const float* Wk = (const float*)d_in[3];
  const float* bk = (const float*)d_in[4];
  const float* Wv = (const float*)d_in[5];
  const float* bv = (const float*)d_in[6];
  const float* Wo = (const float*)d_in[7];
  const float* bo = (const float*)d_in[8];

  const size_t XN = (size_t)M_ * D_;   // 8,388,608
  const size_t WN = (size_t)D_ * D_;   // 4,194,304
  uint16_t* Xb  = (uint16_t*)d_ws;
  uint16_t* Wqb = Xb  + XN;
  uint16_t* Wkb = Wqb + WN;
  uint16_t* Wvb = Wkb + WN;
  uint16_t* Wob = Wvb + WN;
  uint16_t* Qb  = Wob + WN;
  uint16_t* Kb  = Qb  + XN;
  uint16_t* Vtb = Kb  + XN;
  uint16_t* Ob  = Vtb + XN;            // total ~112 MB bf16 workspace

  cvt_f32_bf16<<<1024, 256, 0, stream>>>(hs, Xb,  (int)XN);
  cvt_f32_bf16<<<512,  256, 0, stream>>>(Wq, Wqb, (int)WN);
  cvt_f32_bf16<<<512,  256, 0, stream>>>(Wk, Wkb, (int)WN);
  cvt_f32_bf16<<<512,  256, 0, stream>>>(Wv, Wvb, (int)WN);
  cvt_f32_bf16<<<512,  256, 0, stream>>>(Wo, Wob, (int)WN);

  gemm32<0><<<512, 256, 0, stream>>>(Xb, Wqb, bq, (void*)Qb);   // Q = X Wq^T + bq
  gemm32<0><<<512, 256, 0, stream>>>(Xb, Wkb, bk, (void*)Kb);   // K = X Wk^T + bk
  gemm32<2><<<512, 256, 0, stream>>>(Xb, Wvb, bv, (void*)Vtb);  // Vt[b,h,d,l]

  attn_fa<<<1024, 128, 0, stream>>>(Qb, Kb, Vtb, Ob);           // causal flash attn

  gemm32<1><<<512, 256, 0, stream>>>(Ob, Wob, bo, d_out);       // out = O Wo^T + bo (f32)
}
